// PointNextBackbone_12816182411308
// MI455X (gfx1250) — compile-verified, matrix operand order checked by reading, not run on hardware
//
#include <hip/hip_runtime.h>

typedef __attribute__((ext_vector_type(16))) _Float16 v16h;
typedef __attribute__((ext_vector_type(8)))  _Float16 v8h;
typedef __attribute__((ext_vector_type(8)))  float    v8f;
typedef __attribute__((ext_vector_type(4)))  unsigned int u32x4;
typedef __attribute__((ext_vector_type(8)))  unsigned int u32x8;

#define BATCH 2
#define NPTS  16384
#define NSAMP 32
#define FOFF  8   // feature block offset (halves) inside an A row: [dp3|0*5|fj...]

__device__ __forceinline__ unsigned lds_offset_of(const void* p) {
  // addrspace(3) ptrtoint -> byte offset from workgroup LDS base
  return (unsigned)(unsigned long long)(__attribute__((address_space(3))) const char*)p;
}

__device__ __forceinline__ void wait_asynccnt0() {
#if __has_builtin(__builtin_amdgcn_s_wait_asynccnt)
  __builtin_amdgcn_s_wait_asynccnt(0);
#else
  asm volatile("s_wait_asynccnt 0x0" ::: "memory");
#endif
}

// ---------------------------------------------------------------- mean/std --
__global__ void meanstd_kernel(const float* __restrict__ pos, float* __restrict__ ms) {
  __shared__ float s[256 * 6];
  int b = blockIdx.x, tid = threadIdx.x;
  const float* p = pos + (size_t)b * NPTS * 3;
  float sx = 0.f, sy = 0.f, sz = 0.f, sxx = 0.f, syy = 0.f, szz = 0.f;
  for (int i = tid; i < NPTS; i += 256) {
    float x = p[i * 3 + 0], y = p[i * 3 + 1], z = p[i * 3 + 2];
    sx += x; sy += y; sz += z; sxx += x * x; syy += y * y; szz += z * z;
  }
  s[tid * 6 + 0] = sx; s[tid * 6 + 1] = sy; s[tid * 6 + 2] = sz;
  s[tid * 6 + 3] = sxx; s[tid * 6 + 4] = syy; s[tid * 6 + 5] = szz;
  __syncthreads();
  for (int st = 128; st > 0; st >>= 1) {
    if (tid < st)
      for (int q = 0; q < 6; q++) s[tid * 6 + q] += s[(tid + st) * 6 + q];
    __syncthreads();
  }
  if (tid == 0) {
    for (int a = 0; a < 3; a++) {
      float mean = s[a] / (float)NPTS;
      float var = (s[3 + a] - (float)NPTS * mean * mean) / (float)(NPTS - 1);
      float sd = sqrtf(fmaxf(var, 0.f));
      ms[b * 6 + a] = mean;
      ms[b * 6 + 3 + a] = fmaxf(sd, 1e-6f);
    }
  }
}

// ----------------------------------------------------- stem (features f16) --
__global__ void stem_kernel(const float* __restrict__ pos, const float* __restrict__ ms,
                            const float* __restrict__ w, const float* __restrict__ g,
                            const float* __restrict__ bb, float* __restrict__ xyzn,
                            _Float16* __restrict__ f0h) {
  int t = blockIdx.x * 256 + threadIdx.x;
  if (t >= BATCH * NPTS) return;
  int b = t / NPTS;
  float xn[3];
  for (int a = 0; a < 3; a++) {
    float v = (pos[(size_t)t * 3 + a] - ms[b * 6 + a]) / ms[b * 6 + 3 + a];
    xn[a] = v;
    xyzn[(size_t)t * 3 + a] = v;
  }
  for (int c = 0; c < 32; c++) {
    float acc = xn[0] * w[c] + xn[1] * w[32 + c] + xn[2] * w[64 + c];
    acc = acc * g[c] + bb[c];
    f0h[(size_t)t * 32 + c] = (_Float16)fmaxf(acc, 0.f);
  }
}

// --------------------------------------------------------------------- FPS --
__global__ void fps_kernel(const float* __restrict__ xyzn, int Nin, int K,
                           int* __restrict__ idx_out, float* __restrict__ dscr) {
  __shared__ float rd[1024];
  __shared__ int   ri[1024];
  __shared__ int   s_last;
  int b = blockIdx.x, tid = threadIdx.x;
  const float* p = xyzn + (size_t)b * Nin * 3;
  float* d = dscr + (size_t)b * Nin;
  for (int i = tid; i < Nin; i += 1024) d[i] = 3.4e38f;
  if (tid == 0) { idx_out[(size_t)b * K] = 0; s_last = 0; }
  __syncthreads();
  int last = 0;
  for (int step = 1; step < K; step++) {
    float lx = p[last * 3 + 0], ly = p[last * 3 + 1], lz = p[last * 3 + 2];
    float bv = -1.f; int bidx = 0;
    for (int i = tid; i < Nin; i += 1024) {
      float dx = p[i * 3 + 0] - lx, dy = p[i * 3 + 1] - ly, dz = p[i * 3 + 2] - lz;
      float dd = dx * dx + dy * dy + dz * dz;
      float dn = fminf(d[i], dd);
      d[i] = dn;
      if (dn > bv) { bv = dn; bidx = i; }
    }
    rd[tid] = bv; ri[tid] = bidx;
    __syncthreads();
    for (int s = 512; s > 0; s >>= 1) {
      if (tid < s) {
        float ov = rd[tid + s]; int oi = ri[tid + s];
        if (ov > rd[tid] || (ov == rd[tid] && oi < ri[tid])) { rd[tid] = ov; ri[tid] = oi; }
      }
      __syncthreads();
    }
    if (tid == 0) { s_last = ri[0]; idx_out[(size_t)b * K + step] = ri[0]; }
    __syncthreads();
    last = s_last;
    __syncthreads();
  }
}

// ------------------------------------------------------------------ gather --
__global__ void gather_kernel(const float* __restrict__ curn, const float* __restrict__ curo,
                              const int* __restrict__ fidx, float* __restrict__ cen,
                              float* __restrict__ oout, int M, int Nin) {
  int t = blockIdx.x * 256 + threadIdx.x;
  if (t >= BATCH * M) return;
  int b = t / M;
  int id = fidx[t];
  for (int a = 0; a < 3; a++) {
    cen[(size_t)t * 3 + a]  = curn[((size_t)b * Nin + id) * 3 + a];
    oout[(size_t)t * 3 + a] = curo[((size_t)b * Nin + id) * 3 + a];
  }
}

// --------------------------------------------- ball query (TDM tile loads) --
__global__ void ballquery_kernel(const float* __restrict__ centers, const float* __restrict__ xyzn,
                                 int M, int Nin, float r2, int* __restrict__ nidx) {
  __shared__ float tile[256 * 3];
  __shared__ float bd[64 * NSAMP];
  __shared__ int   bi[64 * NSAMP];
  int bpB = M / 64;
  int b = blockIdx.x / bpB;
  int m = (blockIdx.x % bpB) * 64 + threadIdx.x;
  int tid = threadIdx.x;
  float cx = centers[((size_t)b * M + m) * 3 + 0];
  float cy = centers[((size_t)b * M + m) * 3 + 1];
  float cz = centers[((size_t)b * M + m) * 3 + 2];
  float* myd = &bd[tid * NSAMP];
  int*   myi = &bi[tid * NSAMP];
  unsigned tile_lds = lds_offset_of(tile);
  float worst = -1.f; int wslot = 0; int cnt = 0;
  for (int j0 = 0; j0 < Nin; j0 += 256) {
    __syncthreads();
    if (tid < 32) {
      // 1-D TDM copy: 768 dwords (256 points x 3) global -> LDS tile
      unsigned long long ga =
          (unsigned long long)(const char*)(xyzn + ((size_t)b * Nin + j0) * 3);
      unsigned galo = __builtin_amdgcn_readfirstlane((unsigned)ga);
      unsigned gahi = __builtin_amdgcn_readfirstlane((unsigned)(ga >> 32));
      unsigned ldsu = __builtin_amdgcn_readfirstlane(tile_lds);
      u32x4 g0;
      g0[0] = 1u;                                   // count=1, user descriptor
      g0[1] = ldsu;                                 // lds_addr
      g0[2] = galo;                                 // global_addr[31:0]
      g0[3] = (gahi & 0x01FFFFFFu) | 0x80000000u;   // global_addr[56:32] | type=2
      u32x8 g1;
      g1[0] = 0x20000u;        // data_size=2 (4B), no mask/pad/iterate
      g1[1] = (768u << 16);    // tensor_dim0 = 768 (low 16 bits)
      g1[2] = (1u << 16);      // tensor_dim0 hi=0 | tensor_dim1 = 1
      g1[3] = (768u << 16);    // tensor_dim1 hi=0 | tile_dim0 = 768
      g1[4] = 1u;              // tile_dim1 = 1, tile_dim2 = 0
      g1[5] = 768u;            // tensor_dim0_stride (low 32)
      g1[6] = 0u;
      g1[7] = 0u;
      asm volatile("tensor_load_to_lds %0, %1" :: "s"(g0), "s"(g1) : "memory");
      __builtin_amdgcn_s_wait_tensorcnt(0);
    }
    __syncthreads();
    for (int jj = 0; jj < 256; jj++) {
      float dx = tile[jj * 3 + 0] - cx;
      float dy = tile[jj * 3 + 1] - cy;
      float dz = tile[jj * 3 + 2] - cz;
      float d2 = dx * dx + dy * dy + dz * dz;
      int j = j0 + jj;
      if (cnt < NSAMP) {
        myd[cnt] = d2; myi[cnt] = j;
        if (d2 > worst) { worst = d2; wslot = cnt; }
        cnt++;
      } else if (d2 < worst) {
        myd[wslot] = d2; myi[wslot] = j;
        worst = -1.f;
        for (int s2 = 0; s2 < NSAMP; s2++)
          if (myd[s2] > worst) { worst = myd[s2]; wslot = s2; }
      }
    }
  }
  float nd = 3.4e38f; int ni = 0;
  for (int s2 = 0; s2 < NSAMP; s2++)
    if (myd[s2] < nd) { nd = myd[s2]; ni = myi[s2]; }
  for (int s2 = 0; s2 < NSAMP; s2++) {
    int id = (myd[s2] <= r2) ? myi[s2] : ni;
    nidx[((size_t)b * M + m) * NSAMP + s2] = id;
  }
}

// ---------------------------------------------- pack weights into B frags --
// A-row layout is [dp(3) | zero(5) | fj(cin) | pad], so weight row k maps:
//   k<3 -> w[k], 3<=k<8 -> 0, 8<=k<8+cin -> w[k-5], else 0
// fragment layout (16-bit B, 32x16): lane<16 -> col=lane, halves K{0..7,16..23};
// lane>=16 -> col=lane-16, halves K{8..15,24..31}
__global__ void packw_kernel(const float* __restrict__ w, _Float16* __restrict__ out,
                             int cin, int Kpad, int cout) {
  int t = blockIdx.x * 256 + threadIdx.x;
  int nkc = Kpad / 32;
  int total = (cout / 16) * nkc * 512;
  if (t >= total) return;
  int h = t & 15;
  int lane = (t >> 4) & 31;
  int rest = t >> 9;
  int kc = rest % nkc;
  int ct = rest / nkc;
  int k = kc * 32 + ((h < 8) ? h : h + 8) + ((lane >= 16) ? 8 : 0);
  int n = ct * 16 + (lane & 15);
  float v = 0.f;
  if (k < 3) v = w[(size_t)k * cout + n];
  else if (k >= FOFF && k < FOFF + cin) v = w[(size_t)(k - FOFF + 3) * cout + n];
  out[t] = (_Float16)v;
}

// ------------------- grouped MLP + max-pool (async gather + WMMA f16/f32) --
__global__ void groupmlp_kernel(const float* __restrict__ xyzn, const float* __restrict__ centers,
                                const _Float16* __restrict__ finh, const int* __restrict__ nidx,
                                const _Float16* __restrict__ wpk, const float* __restrict__ gsc,
                                const float* __restrict__ bsc, _Float16* __restrict__ fouth,
                                int Nin, int M, int cin, int cout, int Kpad) {
  extern __shared__ char smem[];
  _Float16* As = (_Float16*)smem;
  int* nbrs = (int*)(smem + (size_t)2 * 32 * Kpad * sizeof(_Float16));
  unsigned lds_base = lds_offset_of(smem);

  int halfM = M >> 1;
  int b = blockIdx.x / halfM;
  int pair = blockIdx.x % halfM;
  int w = threadIdx.x >> 5;
  int lane = threadIdx.x & 31;
  int m = pair * 2 + w;
  size_t cm = (size_t)b * M + m;

  nbrs[w * NSAMP + lane] = nidx[cm * NSAMP + lane];
  __syncthreads();

  int waveoff = w * 32 * Kpad;
  int r = lane;                       // one A row (neighbor) per lane
  int nb = nbrs[w * NSAMP + r];

  // dp header (8 halves, 16B aligned) + zero tail pad
  {
    const float* xs = xyzn + (size_t)(b * Nin + nb) * 3;
    float c0 = centers[cm * 3 + 0], c1 = centers[cm * 3 + 1], c2 = centers[cm * 3 + 2];
    v8h hdr = {};
    hdr[0] = (_Float16)(xs[0] - c0);
    hdr[1] = (_Float16)(xs[1] - c1);
    hdr[2] = (_Float16)(xs[2] - c2);
    *(v8h*)(&As[waveoff + r * Kpad]) = hdr;
    for (int c = FOFF + cin; c < Kpad; c += 8)
      *(v8h*)(&As[waveoff + r * Kpad + c]) = (v8h){};
  }

  // per-lane async DMA of the f16 feature row straight into LDS (ASYNCcnt)
  {
    unsigned long long ga =
        (unsigned long long)(const char*)(finh + (size_t)(b * Nin + nb) * cin);
    unsigned ldso = lds_base + (unsigned)(waveoff + r * Kpad + FOFF) * 2u;
    int nchunk = cin >> 3;  // 16B chunks
    for (int c4 = 0; c4 < nchunk; c4++) {
      asm volatile("global_load_async_to_lds_b128 %0, %1, off"
                   :: "v"(ldso + (unsigned)c4 * 16u),
                      "v"(ga + (unsigned long long)c4 * 16u)
                   : "memory");
    }
  }
  wait_asynccnt0();
  __syncthreads();

  int nkc = Kpad >> 5;
  int nct = cout >> 4;
  int rlo = lane & 15;
  int khi = (lane >= 16) ? 8 : 0;
  union AF { v16h v; unsigned int u[8]; };

  for (int ct = 0; ct < nct; ct++) {
    if (ct + 1 < nct)  // prefetch next column tile's weight fragments
      __builtin_prefetch(wpk + (((size_t)(ct + 1) * nkc) * 32 + lane) * 16, 0, 0);
    v8f acc0 = {};
    v8f acc1 = {};
    for (int kc = 0; kc < nkc; kc++) {
      v16h bf = *(const v16h*)(wpk + (((size_t)ct * nkc + kc) * 32 + lane) * 16);
      AF a0, a1;
      int base0 = waveoff + rlo * Kpad + kc * 32 + khi;
      int base1 = waveoff + (rlo + 16) * Kpad + kc * 32 + khi;
#pragma unroll
      for (int p2 = 0; p2 < 8; p2++) {
        int k2 = (p2 < 4) ? (2 * p2) : (2 * p2 + 8);
        a0.u[p2] = *(const unsigned int*)(&As[base0 + k2]);
        a1.u[p2] = *(const unsigned int*)(&As[base1 + k2]);
      }
      acc0 = __builtin_amdgcn_wmma_f32_16x16x32_f16(false, a0.v, false, bf, (short)0, acc0, false, false);
      acc1 = __builtin_amdgcn_wmma_f32_16x16x32_f16(false, a1.v, false, bf, (short)0, acc1, false, false);
    }
    int col = ct * 16 + rlo;
    float gg = gsc[col], bb2 = bsc[col];
    float mx = 0.f;  // ReLU outputs are >= 0
#pragma unroll
    for (int v2 = 0; v2 < 8; v2++) {
      float h0 = fmaxf(acc0[v2] * gg + bb2, 0.f);
      float h1 = fmaxf(acc1[v2] * gg + bb2, 0.f);
      mx = fmaxf(mx, fmaxf(h0, h1));
    }
    mx = fmaxf(mx, __shfl_xor(mx, 16, 32));
    if (lane < 16) fouth[cm * cout + col] = (_Float16)mx;
  }
}

// ---------------------------------------------------------------- finalize --
__global__ void finalize_kernel(const float* __restrict__ curo, const _Float16* __restrict__ f4h,
                                float* __restrict__ out) {
  int t = blockIdx.x * blockDim.x + threadIdx.x;
  if (t < BATCH * 64 * 3) out[t] = curo[t];
  if (t < BATCH * 512 * 64) {
    int b = t / (512 * 64);
    int rem = t - b * 512 * 64;
    int c = rem >> 6;
    int mm = rem & 63;
    out[BATCH * 64 * 3 + t] = (float)f4h[((size_t)(b * 64 + mm)) * 512 + c];
  }
}

// -------------------------------------------------------------------- host --
extern "C" void kernel_launch(void* const* d_in, const int* in_sizes, int n_in,
                              void* d_out, int out_size, void* d_ws, size_t ws_size,
                              hipStream_t stream) {
  (void)in_sizes; (void)n_in; (void)out_size; (void)ws_size;
  const float* pos    = (const float*)d_in[0];
  const float* stem_w = (const float*)d_in[1];
  const float* stem_g = (const float*)d_in[2];
  const float* stem_b = (const float*)d_in[3];

  char* base = (char*)d_ws;
  size_t off = 0;
  auto bump = [&](size_t bytes) -> char* {
    char* p = base + off;
    off = (off + bytes + 255) & ~(size_t)255;
    return p;
  };
  float*    ms     = (float*)bump((size_t)BATCH * 6 * 4);
  float*    xyzn   = (float*)bump((size_t)BATCH * NPTS * 3 * 4);
  _Float16* f0h    = (_Float16*)bump((size_t)BATCH * NPTS * 32 * 2);
  _Float16* featAh = (_Float16*)bump((size_t)BATCH * 4096 * 64 * 2);
  _Float16* featBh = (_Float16*)bump((size_t)BATCH * 4096 * 64 * 2);
  float*    curnA  = (float*)bump((size_t)BATCH * 4096 * 3 * 4);
  float*    curnB  = (float*)bump((size_t)BATCH * 4096 * 3 * 4);
  float*    curoA  = (float*)bump((size_t)BATCH * 4096 * 3 * 4);
  float*    curoB  = (float*)bump((size_t)BATCH * 4096 * 3 * 4);
  int*      fpsidx = (int*)bump((size_t)BATCH * 4096 * 4);
  int*      nidxb  = (int*)bump((size_t)BATCH * 4096 * NSAMP * 4);
  float*    dscr   = (float*)bump((size_t)BATCH * NPTS * 4);
  _Float16* wpk    = (_Float16*)bump((size_t)288 * 512 * 2);

  meanstd_kernel<<<dim3(BATCH), dim3(256), 0, stream>>>(pos, ms);
  stem_kernel<<<dim3((BATCH * NPTS + 255) / 256), dim3(256), 0, stream>>>(
      pos, ms, stem_w, stem_g, stem_b, xyzn, f0h);

  const int   Nin_a[4]  = {16384, 4096, 1024, 256};
  const int   cin_a[4]  = {32, 64, 128, 256};
  const int   cout_a[4] = {64, 128, 256, 512};
  const int   Kpad_a[4] = {64, 96, 160, 288};
  const float r2_a[4]   = {0.04f, 0.16f, 0.64f, 2.56f};

  const float*    curn_in = xyzn;
  const float*    curo_in = pos;
  const _Float16* f_in    = f0h;
  float*    curn_b[2] = {curnA, curnB};
  float*    curo_b[2] = {curoA, curoB};
  _Float16* feat_b[2] = {featAh, featBh};

  for (int s = 0; s < 4; s++) {
    int Nin = Nin_a[s], M = Nin / 4;
    int cin = cin_a[s], cout = cout_a[s], Kpad = Kpad_a[s];
    const float* w  = (const float*)d_in[4 + 3 * s];
    const float* g  = (const float*)d_in[5 + 3 * s];
    const float* bb = (const float*)d_in[6 + 3 * s];
    float*    cen  = curn_b[s & 1];
    float*    oout = curo_b[s & 1];
    _Float16* fo   = feat_b[s & 1];

    fps_kernel<<<dim3(BATCH), dim3(1024), 0, stream>>>(curn_in, Nin, M, fpsidx, dscr);
    gather_kernel<<<dim3((BATCH * M + 255) / 256), dim3(256), 0, stream>>>(
        curn_in, curo_in, fpsidx, cen, oout, M, Nin);
    ballquery_kernel<<<dim3(BATCH * (M / 64)), dim3(64), 0, stream>>>(
        cen, curn_in, M, Nin, r2_a[s], nidxb);
    {
      int nkc = Kpad / 32;
      int total = (cout / 16) * nkc * 512;
      packw_kernel<<<dim3((total + 255) / 256), dim3(256), 0, stream>>>(
          w, wpk, cin, Kpad, cout);
    }
    size_t smemB = (size_t)2 * 32 * Kpad * sizeof(_Float16) + 2 * NSAMP * sizeof(int);
    groupmlp_kernel<<<dim3(BATCH * (M / 2)), dim3(64), smemB, stream>>>(
        curn_in, cen, f_in, nidxb, wpk, g, bb, fo, Nin, M, cin, cout, Kpad);

    curn_in = cen;
    curo_in = oout;
    f_in = fo;
  }

  finalize_kernel<<<dim3(256), dim3(256), 0, stream>>>(curo_in, f_in, (float*)d_out);
}
